// CNNAttention_31198642438156
// MI455X (gfx1250) — matrix operand order repaired and checked by reading, not compiled, not run on hardware
//
#include <hip/hip_runtime.h>
#include <hip/hip_bf16.h>

// ---------------------------------------------------------------------------
// CNN attention (heads=8, dim_head=64) on MI455X (gfx1250, wave32).
// All GEMMs on v_wmma_f32_16x16x32_bf16 (fp32 accum). QKV conv stages its
// operand tiles into LDS with the Tensor Data Mover (tensor_load_to_lds,
// TENSORcnt double-buffering); waves consume fragments via ds_load_b128.
// Fragments are preloaded in batches so dep-counter waits are graded instead
// of full drains (DS/VMEM return in-order per wave).
// ---------------------------------------------------------------------------

typedef __attribute__((ext_vector_type(16))) __bf16 v16bf;
typedef __attribute__((ext_vector_type(8)))  __bf16 v8bf;
typedef __attribute__((ext_vector_type(8)))  float  v8f;

typedef __attribute__((ext_vector_type(4))) unsigned int tdm_g0_t;
typedef __attribute__((ext_vector_type(8))) int          tdm_g1_t;
typedef __attribute__((ext_vector_type(4))) int          tdm_g23_t;
typedef __attribute__((ext_vector_type(8))) int          tdm_g4_t;

constexpr int   BATCH = 8, CH = 512, HH = 32, WW = 32, NPIX = 1024;
constexpr int   G = 8, DH = 64, INNER = 512, OC3 = 1536;
constexpr float SMOOTH = 1e-4f, BN_EPS = 1e-5f;

// ---- workspace layout (bytes, all 256-aligned) ----------------------------
constexpr size_t OFF_XP  = 0;                                   // bf16 [8][34][34][512]
constexpr size_t SZ_XP   = (size_t)8 * 34 * 34 * 512 * 2;
constexpr size_t OFF_WQT = OFF_XP + SZ_XP;                      // bf16 [9][1536][512]
constexpr size_t SZ_WQT  = (size_t)9 * 1536 * 512 * 2;
constexpr size_t OFF_QH  = OFF_WQT + SZ_WQT;                    // bf16 [64][1024][64]
constexpr size_t SZ_QH   = (size_t)64 * 1024 * 64 * 2;
constexpr size_t OFF_KH  = OFF_QH + SZ_QH;                      // bf16 [64][1024][64]
constexpr size_t OFF_VT  = OFF_KH + SZ_QH;                      // bf16 [64][64][1024]  (V transposed)
constexpr size_t OFF_QN  = OFF_VT + SZ_QH;                      // f32  [64][1024]
constexpr size_t SZ_NRM  = (size_t)64 * 1024 * 4;
constexpr size_t OFF_KN  = OFF_QN + SZ_NRM;                     // f32  [64][1024]
constexpr size_t OFF_RS  = OFF_KN + SZ_NRM;                     // f32  [8][1024]  dmask row sums
constexpr size_t OFF_OA  = OFF_RS + (size_t)8 * 1024 * 4;       // bf16 [8192][512] attn output
constexpr size_t OFF_WO  = OFF_OA + (size_t)8192 * 512 * 2;     // bf16 [512][512]
constexpr size_t OFF_Y   = OFF_WO + (size_t)512 * 512 * 2;      // f32  [8192][512] pre-BN
constexpr size_t OFF_MV  = OFF_Y + (size_t)8192 * 512 * 4;      // f32  [1024] mean|var

// ---- helpers --------------------------------------------------------------
static __device__ __forceinline__ unsigned short f2bf(float f) {
    unsigned u = __float_as_uint(f);
    unsigned r = u + 0x7FFFu + ((u >> 16) & 1u);   // RNE
    return (unsigned short)(r >> 16);
}
static __device__ __forceinline__ float bf2f(unsigned short u) {
    return __uint_as_float(((unsigned)u) << 16);
}
// low 32 bits of an LDS-aperture flat address == LDS byte offset (ISA 10.2)
static __device__ __forceinline__ unsigned lds_off(const void* p) {
    return (unsigned)(unsigned long long)p;
}
// assemble a 16x32-bf16 wave fragment half from two 16-byte chunks
static __device__ __forceinline__ v16bf mk16(const unsigned short* lo,
                                             const unsigned short* hi) {
    v8bf a = *(const v8bf*)lo;
    v8bf b = *(const v8bf*)hi;
    v16bf r;
#pragma unroll
    for (int i = 0; i < 8; ++i) { r[i] = a[i]; r[i + 8] = b[i]; }
    return r;
}
static __device__ __forceinline__ v8f wmma_bf16(v16bf a, v16bf b, v8f c) {
    return __builtin_amdgcn_wmma_f32_16x16x32_bf16(false, a, false, b,
                                                   (short)0, c, false, false);
}
static __device__ __forceinline__ v8f v8f_zero() {
    v8f z = {0.f, 0.f, 0.f, 0.f, 0.f, 0.f, 0.f, 0.f};
    return z;
}

// ---- Tensor Data Mover: build D# (ISA §8.3-8.6) and issue a load -----------
// 3D tile: dim0 (contiguous, elements) x dim1 (stride0) x dim2 (stride1);
// data_size = 2 bytes (bf16). tile2==0 -> plain 2D tile.
// This toolchain exposes the 6-arg builtin:
//   (uint32x4 g0, int32x8 g1, int32x4 g2, int32x4 g3, int32x8 g4, i32 cpol)
static __device__ __forceinline__ void tdm_load(
    unsigned lds_byte, unsigned long long gaddr,
    unsigned tdim0, unsigned tdim1, unsigned tdim2,
    unsigned tile0, unsigned tile1, unsigned tile2,
    unsigned long long stride0, unsigned long long stride1)
{
    tdm_g0_t g0;
    g0[0] = 1u;                                                  // count=1, user D#
    g0[1] = lds_byte;                                            // lds_addr
    g0[2] = (unsigned)(gaddr & 0xFFFFFFFFull);                   // global_addr lo
    g0[3] = (unsigned)((gaddr >> 32) & 0x1FFFFFFull) | (2u << 30); // addr hi | type=2
    tdm_g1_t g1;
    g1[0] = (int)(1u << 16);                                     // data_size=1 (2B)
    g1[1] = (int)((tdim0 & 0xFFFFu) << 16);                      // tensor_dim0 lo16
    g1[2] = (int)((tdim0 >> 16) | ((tdim1 & 0xFFFFu) << 16));    // dim0 hi | dim1 lo
    g1[3] = (int)((tdim1 >> 16) | (tile0 << 16));                // dim1 hi | tile_dim0
    g1[4] = (int)(tile1 | (tile2 << 16));                        // tile_dim1 | tile_dim2
    g1[5] = (int)(stride0 & 0xFFFFFFFFull);                      // dim0_stride lo32
    g1[6] = (int)(((stride0 >> 32) & 0xFFFFull) |
                  ((stride1 & 0xFFFFull) << 16));                // dim0 hi16 | dim1 lo16
    g1[7] = (int)((stride1 >> 16) & 0xFFFFFFFFull);              // dim1_stride [47:16]
    tdm_g23_t g2;
    g2[0] = (int)tdim2; g2[1] = 0; g2[2] = 0; g2[3] = 0;         // tensor_dim2, rest 0
    tdm_g23_t g3;
    g3[0] = 0; g3[1] = 0; g3[2] = 0; g3[3] = 0;
    tdm_g4_t g4;
    g4[0] = 0; g4[1] = 0; g4[2] = 0; g4[3] = 0;
    g4[4] = 0; g4[5] = 0; g4[6] = 0; g4[7] = 0;
    __builtin_amdgcn_tensor_load_to_lds(g0, g1, g2, g3, g4, 0);
}

// ---------------------------------------------------------------------------
// Pack kernels: fp32 NCHW -> bf16 layouts tuned for WMMA fragment loads
// ---------------------------------------------------------------------------
__global__ void k_pack_x(const float* __restrict__ x, unsigned short* __restrict__ xp) {
    long idx = (long)blockIdx.x * blockDim.x + threadIdx.x;
    const long total = (long)8 * 34 * 34 * 512;
    if (idx >= total) return;
    int c = (int)(idx & 511);
    long t = idx >> 9;
    int xx = (int)(t % 34); t /= 34;
    int yy = (int)(t % 34); t /= 34;
    int b  = (int)t;
    unsigned short v = 0;
    if (yy >= 1 && yy <= 32 && xx >= 1 && xx <= 32)
        v = f2bf(x[(((long)b * CH + c) * HH + (yy - 1)) * WW + (xx - 1)]);
    xp[idx] = v;
}

__global__ void k_pack_wq(const float* __restrict__ wq, unsigned short* __restrict__ wqt) {
    long idx = (long)blockIdx.x * blockDim.x + threadIdx.x;
    const long total = (long)9 * 1536 * 512;
    if (idx >= total) return;
    int c = (int)(idx & 511);
    long t = idx >> 9;
    int o = (int)(t % 1536);
    int p = (int)(t / 1536);
    int kh = p / 3, kw = p % 3;
    wqt[idx] = f2bf(wq[(((long)o * CH + c) * 3 + kh) * 3 + kw]);
}

__global__ void k_pack_wo(const float* __restrict__ wo, unsigned short* __restrict__ wob) {
    int idx = blockIdx.x * blockDim.x + threadIdx.x;
    if (idx >= 512 * 512) return;
    wob[idx] = f2bf(wo[idx]);   // w_out is already [Cout][K]
}

// ---------------------------------------------------------------------------
// QKV 3x3 conv as implicit GEMM: M=8192 pixels, N=1536, K=9*512.
// Workgroup tile 64(M) x 128(N); 8 waves in 4x2. Operand tiles staged into
// LDS by the TDM (double buffered), consumed as WMMA fragments via ds loads.
// ---------------------------------------------------------------------------
__global__ void __launch_bounds__(256)
k_conv_qkv(const unsigned short* __restrict__ xp, const unsigned short* __restrict__ wqt,
           unsigned short* __restrict__ qh, unsigned short* __restrict__ khm,
           unsigned short* __restrict__ vt) {
    __shared__ __align__(16) unsigned short sA[2][64][32];    //  8 KB: 64 pixels x 32 c
    __shared__ __align__(16) unsigned short sB[2][128][32];   // 16 KB: 128 outs x 32 c

    const int tid = threadIdx.x;
    const int wvL = tid >> 5;                 // 0..7
    const int l   = tid & 31;
    const int hf  = l >> 4, lan = l & 15;
    const int mi  = wvL & 3;                  // 4 M-subtiles of 16 pixels
    const int ni  = wvL >> 2;                 // 2 N-subtiles of 64 channels

    const int mt = blockIdx.x & 127;          // 128 M-tiles of 64 pixels
    const int nt = blockIdx.x >> 7;           // 12 N-tiles of 128 channels
    const int m0 = mt * 64;
    const int b  = m0 >> 10;
    const int y0 = (m0 & 1023) >> 5;          // first of two image rows
    const int o0 = nt * 128;

    const unsigned long long xbase = (unsigned long long)xp;
    const unsigned long long wbase = (unsigned long long)wqt;
    const unsigned ldsA[2] = {lds_off(&sA[0][0][0]), lds_off(&sA[1][0][0])};
    const unsigned ldsB[2] = {lds_off(&sB[0][0][0]), lds_off(&sB[1][0][0])};

    const int NCHUNK = 9 * 16;                // (kh,kw) x c-chunks of 32

    // issue chunk 0 (wave 0 only; TDM ignores EXEC, one instruction per wave)
    if (wvL == 0) {
        // chunk 0: p=0 (kh=0,kw=0), cb=0
        unsigned long long ga =
            xbase + (((unsigned long long)b * 34 + (y0 + 0)) * 34 + 0) * 512ull * 2ull;
        tdm_load(ldsA[0], ga, 32, 32, 2, 32, 32, 2, 512ull, 34ull * 512ull);
        unsigned long long gb =
            wbase + (((unsigned long long)0 * 1536 + o0) * 512ull) * 2ull;
        tdm_load(ldsB[0], gb, 32, 128, 0, 32, 128, 0, 512ull, 0ull);
    }

    v8f acc[4] = {v8f_zero(), v8f_zero(), v8f_zero(), v8f_zero()};

    for (int k = 0; k < NCHUNK; ++k) {
        const int buf = k & 1;
        if (wvL == 0) __builtin_amdgcn_s_wait_tensorcnt(0);   // buf ready
        __syncthreads();                                       // all waves see it
        if (wvL == 0 && (k + 1) < NCHUNK) {                    // prefetch next chunk
            const int kn = k + 1;
            const int p  = kn >> 4;
            const int cb = (kn & 15) * 32;
            const int kh = p / 3, kw = p % 3;
            unsigned long long ga = xbase +
                ((((unsigned long long)b * 34 + (y0 + kh)) * 34 + kw) * 512ull + cb) * 2ull;
            tdm_load(ldsA[buf ^ 1], ga, 32, 32, 2, 32, 32, 2, 512ull, 34ull * 512ull);
            unsigned long long gb = wbase +
                ((((unsigned long long)p * 1536 + o0) * 512ull) + cb) * 2ull;
            tdm_load(ldsB[buf ^ 1], gb, 32, 128, 0, 32, 128, 0, 512ull, 0ull);
        }
        // compute current chunk from LDS: issue ALL fragment loads first so
        // the dscnt waits are graded (DS returns in order within a wave),
        // then run the 4 WMMAs against in-flight tail loads.
        const int pl = mi * 16 + lan;                          // pixel row in tile
        v16bf a = mk16(&sA[buf][pl][hf * 16], &sA[buf][pl][hf * 16 + 8]);
        v16bf bfr[4];
#pragma unroll
        for (int s = 0; s < 4; ++s) {
            const int ol = ni * 64 + s * 16 + lan;             // out-channel in tile
            bfr[s] = mk16(&sB[buf][ol][hf * 16], &sB[buf][ol][hf * 16 + 8]);
        }
#pragma unroll
        for (int s = 0; s < 4; ++s)
            acc[s] = wmma_bf16(a, bfr[s], acc[s]);
    }

#pragma unroll
    for (int s = 0; s < 4; ++s) {
        int o = o0 + ni * 64 + s * 16 + lan;     // output channel (D col = lane&15)
        int which = o >> 9;                      // 0=q 1=k 2=v
        int oc = o & 511;
        int g = oc >> 6, dd = oc & 63;
        long bg = (long)b * 8 + g;
#pragma unroll
        for (int r = 0; r < 8; ++r) {
            int n = (m0 & 1023) + mi * 16 + r + 8 * hf;   // D row = vgpr + 8*(lane>=16)
            unsigned short v = f2bf(acc[s][r]);
            if (which == 0)      qh[(bg * 1024 + n) * 64 + dd] = v;
            else if (which == 1) khm[(bg * 1024 + n) * 64 + dd] = v;
            else                 vt[(bg * 64 + dd) * 1024 + n] = v;
        }
    }
}

// ---------------------------------------------------------------------------
// Row norms of Q/K (cosine denominator)
// ---------------------------------------------------------------------------
__global__ void k_norms(const unsigned short* __restrict__ qh,
                        const unsigned short* __restrict__ khm,
                        float* __restrict__ qn, float* __restrict__ kn) {
    int t = blockIdx.x * blockDim.x + threadIdx.x;
    if (t >= 64 * 1024) return;
    const unsigned short* q = qh + (long)t * 64;
    const unsigned short* k = khm + (long)t * 64;
    float sq = 0.f, sk = 0.f;
#pragma unroll 8
    for (int i = 0; i < 64; ++i) {
        float a = bf2f(q[i]); sq += a * a;
        float c = bf2f(k[i]); sk += c * c;
    }
    qn[t] = sqrtf(sq + SMOOTH);
    kn[t] = sqrtf(sk + SMOOTH);
}

// ---------------------------------------------------------------------------
// Per-head Gaussian distance-mask row sums (for normalization)
// ---------------------------------------------------------------------------
__global__ void k_rowsum(const float* __restrict__ headsita, float* __restrict__ rs) {
    int t = blockIdx.x * blockDim.x + threadIdx.x;
    if (t >= 8 * 1024) return;
    int g = t >> 10, n = t & 1023;
    float th  = headsita[g];
    float sig = 1.f / (1.f + __expf(-th));
    float sg  = sig * (0.4f - 0.003f) + 0.003f;
    float fac = 1.f / (2.f * sg * sg);
    int yn = n >> 5, xn = n & 31;
    float sum = 0.f;
    for (int m = 0; m < 1024; ++m) {
        int ym = m >> 5, xm = m & 31;
        float dy = (float)(yn - ym) * (1.f / 32.f);
        float dx = (float)(xn - xm) * (1.f / 32.f);
        sum += __expf(-fac * (dy * dy + dx * dx));
    }
    rs[t] = sum;
}

// ---------------------------------------------------------------------------
// Attention: per wave one 16-query block of one (b,g).
// S = QK^T (WMMA) -> scale by cosine norm * dmask -> bf16 via LDS re-layout ->
// O += S * V (WMMA).  All 4 waves in a block share trip counts (barrier-safe).
// V fragments are prefetched before the LDS re-layout so the global loads
// overlap the exp/scale VALU work.
// ---------------------------------------------------------------------------
__global__ void __launch_bounds__(128)
k_attn(const unsigned short* __restrict__ qh, const unsigned short* __restrict__ khm,
       const unsigned short* __restrict__ vt, const float* __restrict__ qn,
       const float* __restrict__ kn, const float* __restrict__ rs,
       const float* __restrict__ headsita, unsigned short* __restrict__ oa) {
    __shared__ __align__(16) unsigned short sS[4][16][32];

    int wvL = threadIdx.x >> 5;
    int wv  = blockIdx.x * 4 + wvL;
    int l   = threadIdx.x & 31, hf = l >> 4, lan = l & 15;
    int b   = wv >> 9;
    int rem = wv & 511;
    int g   = rem >> 6;
    int rb  = (rem & 63) * 16;
    long bg = (long)b * 8 + g;

    float th  = headsita[g];
    float sig = 1.f / (1.f + __expf(-th));
    float sg  = sig * (0.4f - 0.003f) + 0.003f;
    float fac = 1.f / (2.f * sg * sg);

    // Q fragments for full d=64 (two K-steps), kept live across the key loop
    const unsigned short* qrow = qh + (bg * 1024 + rb + lan) * 64;
    v16bf aq0 = mk16(qrow + hf * 16,      qrow + hf * 16 + 8);
    v16bf aq1 = mk16(qrow + 32 + hf * 16, qrow + 32 + hf * 16 + 8);

    float qn8[8], irs8[8];
    int   qy8[8], qx8[8];
#pragma unroll
    for (int r = 0; r < 8; ++r) {
        int qr = rb + r + 8 * hf;
        qn8[r] = qn[bg * 1024 + qr];
        irs8[r] = 1.f / rs[(long)g * 1024 + qr];
        qy8[r] = qr >> 5; qx8[r] = qr & 31;
    }

    v8f acc[4] = {v8f_zero(), v8f_zero(), v8f_zero(), v8f_zero()};

    for (int mb = 0; mb < 1024; mb += 32) {
        // prefetch this chunk's V fragments early (independent of LDS stage)
        v16bf bv[4];
#pragma unroll
        for (int s = 0; s < 4; ++s) {
            int dd = s * 16 + lan;
            const unsigned short* vrow = vt + (bg * 64 + dd) * 1024 + mb + hf * 16;
            bv[s] = mk16(vrow, vrow + 8);
        }
#pragma unroll
        for (int t = 0; t < 2; ++t) {
            int key = mb + t * 16 + lan;
            const unsigned short* krow = khm + (bg * 1024 + key) * 64;
            v16bf b0 = mk16(krow + hf * 16,      krow + hf * 16 + 8);
            v16bf b1 = mk16(krow + 32 + hf * 16, krow + 32 + hf * 16 + 8);
            v8f s0 = v8f_zero();
            s0 = wmma_bf16(aq0, b0, s0);
            s0 = wmma_bf16(aq1, b1, s0);
            float knv = kn[bg * 1024 + key];
            int ky = key >> 5, kx = key & 31;
#pragma unroll
            for (int r = 0; r < 8; ++r) {
                float dy = (float)(qy8[r] - ky) * (1.f / 32.f);
                float dx = (float)(qx8[r] - kx) * (1.f / 32.f);
                float dm = __expf(-fac * (dy * dy + dx * dx)) * irs8[r];
                float v  = s0[r] / (qn8[r] * knv + SMOOTH) * dm;
                sS[wvL][r + 8 * hf][t * 16 + lan] = f2bf(v);   // [query][key-in-32]
            }
        }
        __syncthreads();
        // re-read as 16x32 A fragment (A layout)
        v16bf a2 = mk16(&sS[wvL][lan][hf * 16], &sS[wvL][lan][hf * 16 + 8]);
        __syncthreads();
#pragma unroll
        for (int s = 0; s < 4; ++s)
            acc[s] = wmma_bf16(a2, bv[s], acc[s]);
    }

#pragma unroll
    for (int s = 0; s < 4; ++s) {
        int dd = s * 16 + lan;
#pragma unroll
        for (int r = 0; r < 8; ++r) {
            int n = rb + r + 8 * hf;
            oa[((long)b * 1024 + n) * 512 + g * 64 + dd] = f2bf(acc[s][r]);
        }
    }
}

// ---------------------------------------------------------------------------
// 1x1 out conv: y[8192,512] = OA[8192,512] * Wo^T   (pre-BN, fp32 out)
// ---------------------------------------------------------------------------
__global__ void __launch_bounds__(256)
k_out_gemm(const unsigned short* __restrict__ oa, const unsigned short* __restrict__ wo,
           float* __restrict__ y) {
    int wv = blockIdx.x * 8 + (threadIdx.x >> 5);
    int l  = threadIdx.x & 31, hf = l >> 4, lan = l & 15;
    int mt = wv & 511;
    int nt = wv >> 9;   // 0..7

    v8f acc[4] = {v8f_zero(), v8f_zero(), v8f_zero(), v8f_zero()};
    const unsigned short* arow = oa + ((long)mt * 16 + lan) * 512;
    for (int jb = 0; jb < 512; jb += 32) {
        __builtin_prefetch(arow + jb + 64, 0, 1);   // -> global_prefetch_b8
        v16bf a = mk16(arow + jb + hf * 16, arow + jb + hf * 16 + 8);
        v16bf bw[4];
#pragma unroll
        for (int s = 0; s < 4; ++s) {
            int c = nt * 64 + s * 16 + lan;
            const unsigned short* wrow = wo + (long)c * 512 + jb + hf * 16;
            bw[s] = mk16(wrow, wrow + 8);
        }
#pragma unroll
        for (int s = 0; s < 4; ++s)
            acc[s] = wmma_bf16(a, bw[s], acc[s]);
    }
#pragma unroll
    for (int s = 0; s < 4; ++s) {
        int c = nt * 64 + s * 16 + lan;
#pragma unroll
        for (int r = 0; r < 8; ++r) {
            int m = mt * 16 + r + 8 * hf;
            y[(long)m * 512 + c] = acc[s][r];
        }
    }
}

// ---------------------------------------------------------------------------
// BatchNorm (batch stats) + ReLU
// ---------------------------------------------------------------------------
__global__ void __launch_bounds__(256)
k_bn_stats(const float* __restrict__ y, float* __restrict__ mv) {
    __shared__ float s1[256], s2[256];
    int c = blockIdx.x;
    float a = 0.f, b = 0.f;
    for (int m = threadIdx.x; m < 8192; m += 256) {
        float v = y[(long)m * 512 + c];
        a += v; b += v * v;
    }
    s1[threadIdx.x] = a; s2[threadIdx.x] = b;
    __syncthreads();
    for (int st = 128; st > 0; st >>= 1) {
        if ((int)threadIdx.x < st) {
            s1[threadIdx.x] += s1[threadIdx.x + st];
            s2[threadIdx.x] += s2[threadIdx.x + st];
        }
        __syncthreads();
    }
    if (threadIdx.x == 0) {
        float mean = s1[0] * (1.f / 8192.f);
        float var  = s2[0] * (1.f / 8192.f) - mean * mean;
        mv[c] = mean; mv[512 + c] = var;
    }
}

__global__ void k_bn_apply(const float* __restrict__ y, const float* __restrict__ mv,
                           const float* __restrict__ gamma, const float* __restrict__ beta,
                           float* __restrict__ out) {
    long idx = (long)blockIdx.x * blockDim.x + threadIdx.x;
    const long total = (long)8 * 512 * 1024;
    if (idx >= total) return;
    int hw = (int)(idx & 1023);
    long t = idx >> 10;
    int c  = (int)(t & 511);
    int b  = (int)(t >> 9);
    int m  = b * 1024 + hw;
    float v = y[(long)m * 512 + c];
    float r = (v - mv[c]) * rsqrtf(mv[512 + c] + BN_EPS) * gamma[c] + beta[c];
    out[idx] = r > 0.f ? r : 0.f;
}

// ---------------------------------------------------------------------------
extern "C" void kernel_launch(void* const* d_in, const int* in_sizes, int n_in,
                              void* d_out, int out_size, void* d_ws, size_t ws_size,
                              hipStream_t stream) {
    const float* x        = (const float*)d_in[0];
    const float* w_qkv    = (const float*)d_in[1];
    const float* headsita = (const float*)d_in[2];
    const float* w_out    = (const float*)d_in[3];
    const float* bn_gamma = (const float*)d_in[4];
    const float* bn_beta  = (const float*)d_in[5];
    float* out = (float*)d_out;

    char* ws = (char*)d_ws;
    unsigned short* XP  = (unsigned short*)(ws + OFF_XP);
    unsigned short* WQT = (unsigned short*)(ws + OFF_WQT);
    unsigned short* QH  = (unsigned short*)(ws + OFF_QH);
    unsigned short* KH  = (unsigned short*)(ws + OFF_KH);
    unsigned short* VT  = (unsigned short*)(ws + OFF_VT);
    float*          QN  = (float*)(ws + OFF_QN);
    float*          KN  = (float*)(ws + OFF_KN);
    float*          RS  = (float*)(ws + OFF_RS);
    unsigned short* OA  = (unsigned short*)(ws + OFF_OA);
    unsigned short* WO  = (unsigned short*)(ws + OFF_WO);
    float*          Y   = (float*)(ws + OFF_Y);
    float*          MV  = (float*)(ws + OFF_MV);

    k_pack_x  <<<18496, 256, 0, stream>>>(x, XP);
    k_pack_wq <<<27648, 256, 0, stream>>>(w_qkv, WQT);
    k_pack_wo <<<1024,  256, 0, stream>>>(w_out, WO);

    k_conv_qkv<<<1536, 256, 0, stream>>>(XP, WQT, QH, KH, VT);

    k_norms  <<<256, 256, 0, stream>>>(QH, KH, QN, KN);
    k_rowsum <<<32,  256, 0, stream>>>(headsita, RS);

    k_attn<<<1024, 128, 0, stream>>>(QH, KH, VT, QN, KN, RS, headsita, OA);

    k_out_gemm<<<512, 256, 0, stream>>>(OA, WO, Y);
    k_bn_stats<<<512, 256, 0, stream>>>(Y, MV);
    k_bn_apply<<<16384, 256, 0, stream>>>(Y, MV, bn_gamma, bn_beta, out);
}
